// FFTInspiredAttentionFused_60378650247739
// MI455X (gfx1250) — matrix operand order repaired and checked by reading, not compiled, not run on hardware
//
#include <hip/hip_runtime.h>
#include <hip/hip_bf16.h>

typedef __attribute__((ext_vector_type(16))) __bf16 v16bf;
typedef __attribute__((ext_vector_type(8)))  float  v8f;

#define DIMF   128
#define NSEQ   8192
#define NBATCH 16
#define LOGN   13
#define QKSCALE 0.17677669529663687f  // 1/sqrt(32)

__device__ __forceinline__ unsigned short f2bf(float f) {
    unsigned int u = __float_as_uint(f);
    unsigned int r = u + 0x7FFFu + ((u >> 16) & 1u);  // round-to-nearest-even
    return (unsigned short)(r >> 16);
}
__device__ __forceinline__ __bf16 bfbits(unsigned short s) {
    return __builtin_bit_cast(__bf16, s);
}

// ---------------------------------------------------------------------------
// Kernel 0: Mt[s][e][d] = sum_c Wk[s][e][c] * Wq[s][d][c]   (bf16 out)
// qk_weights layout: (16, 2, 128, 128); [s][0]=Wq, [s][1]=Wk
// ---------------------------------------------------------------------------
__global__ __launch_bounds__(256) void precompute_mt_kernel(
    const float* __restrict__ qkw, unsigned short* __restrict__ mt)
{
    int gid = blockIdx.x * 256 + threadIdx.x;   // exactly 13*16384 threads
    int s   = gid >> 14;
    int rem = gid & 16383;
    int e   = rem >> 7;
    int d   = rem & 127;
    const float* wq = qkw + (((long)s * 2 + 0) * DIMF + d) * DIMF;
    const float* wk = qkw + (((long)s * 2 + 1) * DIMF + e) * DIMF;
    float acc = 0.f;
    #pragma unroll 8
    for (int c = 0; c < DIMF; ++c) acc += wk[c] * wq[c];
    mt[gid] = f2bf(acc);
}

// ---------------------------------------------------------------------------
// Per-stage fused butterfly kernel.
//   u = x_a @ M   via v_wmma_f32_16x16x32_bf16   (M^T rows staged in LDS)
//   s_aa = u . x_a ; s_ab = u . x_b  (f32, from C-fragment layout)
//   2-way softmax, in-place mix of v (= d_out)
// One wave32 per 16 butterfly pairs. 8 waves / block, 512 blocks.
// ---------------------------------------------------------------------------
__global__ __launch_bounds__(256) void butterfly_stage_kernel(
    const float* __restrict__ x, const unsigned short* __restrict__ mt,
    float* __restrict__ v, int stride)
{
    __shared__ unsigned short MtLds[128][136];   // 272B row pitch (pad: bank spread)
    __shared__ float w0s[8][16];
    __shared__ float w1s[8][16];

    const int tid = threadIdx.x;

    // Stage Mt (128x128 bf16 = 32KB) into LDS, 16B chunks
    for (int i = tid; i < 2048; i += 256) {
        int row = i >> 4;
        int col = (i & 15) * 8;
        *(uint4*)(&MtLds[row][col]) = *(const uint4*)(mt + row * DIMF + col);
    }
    __syncthreads();

    const int lane = tid & 31;
    const int wave = tid >> 5;
    const int lnm  = lane & 15;
    const int half = lane >> 4;

    const int tileId = blockIdx.x * 8 + wave;       // 4096 tiles total
    const int b      = tileId >> 8;                 // 256 tiles per batch
    const int a0     = (tileId & 255) * 16;         // base a-index (0..4095)
    const long bBase = (long)b * NSEQ;
    const int  sm1   = stride - 1;

    // ---- A fragments: 16 'a' rows of x, bf16, ISA 16-bit A layout ----
    int rowA_frag;
    { int al = a0 + lnm; rowA_frag = ((al & ~sm1) << 1) | (al & sm1); }
    const float* xrow = x + (bBase + rowA_frag) * DIMF;

    v16bf afr[4];
    #pragma unroll
    for (int kk = 0; kk < 4; ++kk) {
        int koff = kk * 32 + half * 8;
        float4 c0 = *(const float4*)(xrow + koff);
        float4 c1 = *(const float4*)(xrow + koff + 4);
        float4 c2 = *(const float4*)(xrow + koff + 16);
        float4 c3 = *(const float4*)(xrow + koff + 20);
        v16bf a;
        a[0]  = (__bf16)c0.x; a[1]  = (__bf16)c0.y; a[2]  = (__bf16)c0.z; a[3]  = (__bf16)c0.w;
        a[4]  = (__bf16)c1.x; a[5]  = (__bf16)c1.y; a[6]  = (__bf16)c1.z; a[7]  = (__bf16)c1.w;
        a[8]  = (__bf16)c2.x; a[9]  = (__bf16)c2.y; a[10] = (__bf16)c2.z; a[11] = (__bf16)c2.w;
        a[12] = (__bf16)c3.x; a[13] = (__bf16)c3.y; a[14] = (__bf16)c3.z; a[15] = (__bf16)c3.w;
        afr[kk] = a;
    }

    // ---- U(16x128) = Xa(16x128) @ M(128x128): 8 col-tiles x 4 K-steps ----
    v8f acc[8] = {};
    #pragma unroll
    for (int ct = 0; ct < 8; ++ct) {
        #pragma unroll
        for (int kk = 0; kk < 4; ++kk) {
            // B layout: lane holds column N=lnm, K = 8*half+e (e<8), 16+8*half+(e-8) (e>=8)
            const unsigned short* mrow = &MtLds[ct * 16 + lnm][kk * 32 + half * 8];
            v16bf bfr;
            #pragma unroll
            for (int e = 0; e < 16; ++e) bfr[e] = bfbits(mrow[e + (e >= 8 ? 8 : 0)]);
            acc[ct] = __builtin_amdgcn_wmma_f32_16x16x32_bf16(
                false, afr[kk], false, bfr, (short)0, acc[ct], false, false);
        }
    }

    // ---- logits from C-fragment layout: lane holds U[r][ct*16+lnm], r=creg+8*half
    #pragma unroll
    for (int creg = 0; creg < 8; ++creg) {
        int r  = creg + half * 8;
        int al = a0 + r;
        int nA = ((al & ~sm1) << 1) | (al & sm1);
        int nB = nA + stride;
        const float* xa = x + (bBase + nA) * DIMF;
        const float* xb = x + (bBase + nB) * DIMF;
        float paa = 0.f, pab = 0.f;
        #pragma unroll
        for (int ct = 0; ct < 8; ++ct) {
            float u = acc[ct][creg];
            int   c = ct * 16 + lnm;
            paa += u * xa[c];
            pab += u * xb[c];
        }
        #pragma unroll
        for (int off = 8; off > 0; off >>= 1) {
            paa += __shfl_xor(paa, off, 16);
            pab += __shfl_xor(pab, off, 16);
        }
        if (lnm == 0) {
            float l0 = paa * QKSCALE, l1 = pab * QKSCALE;
            float m  = fmaxf(l0, l1);
            float e0 = __expf(l0 - m), e1 = __expf(l1 - m);
            float inv = 1.f / (e0 + e1);
            w0s[wave][r] = e0 * inv;
            w1s[wave][r] = e1 * inv;
        }
    }
    __syncthreads();

    // ---- in-place mix of v: out_a = w0*va + w1*vb ; out_b = w1*va + w0*vb
    const float4* vin  = (const float4*)v;
    float4*       vout = (float4*)v;
    #pragma unroll 4
    for (int q4 = lane; q4 < 512; q4 += 32) {   // 16 rows * 32 float4
        int r  = q4 >> 5;
        int c4 = q4 & 31;
        int al = a0 + r;
        int nA = ((al & ~sm1) << 1) | (al & sm1);
        int nB = nA + stride;
        long ia = (bBase + nA) * 32 + c4;
        long ib = (bBase + nB) * 32 + c4;
        float4 va = vin[ia];
        float4 vb = vin[ib];
        float w0 = w0s[wave][r], w1 = w1s[wave][r];
        float4 oa, ob;
        oa.x = w0 * va.x + w1 * vb.x;  ob.x = w1 * va.x + w0 * vb.x;
        oa.y = w0 * va.y + w1 * vb.y;  ob.y = w1 * va.y + w0 * vb.y;
        oa.z = w0 * va.z + w1 * vb.z;  ob.z = w1 * va.z + w0 * vb.z;
        oa.w = w0 * va.w + w1 * vb.w;  ob.w = w1 * va.w + w0 * vb.w;
        vout[ia] = oa;
        vout[ib] = ob;
    }
}

// ---------------------------------------------------------------------------
extern "C" void kernel_launch(void* const* d_in, const int* in_sizes, int n_in,
                              void* d_out, int out_size, void* d_ws, size_t ws_size,
                              hipStream_t stream)
{
    const float* x   = (const float*)d_in[0];   // (16, 8192, 128) f32
    const float* qkw = (const float*)d_in[1];   // (16, 2, 128, 128) f32
    float* out = (float*)d_out;                 // (16, 8192, 128) f32, doubles as v
    unsigned short* mt = (unsigned short*)d_ws; // 13 * 128*128 bf16 = 416 KB

    // v <- x
    hipMemcpyAsync(out, x, (size_t)NBATCH * NSEQ * DIMF * sizeof(float),
                   hipMemcpyDeviceToDevice, stream);

    // Mt_s = Wk_s @ Wq_s^T for all 13 stages
    precompute_mt_kernel<<<(LOGN * 16384) / 256, 256, 0, stream>>>(qkw, mt);

    // 13 butterfly stages, in-place on `out`
    for (int s = 0; s < LOGN; ++s) {
        butterfly_stage_kernel<<<512, 256, 0, stream>>>(
            x, mt + (size_t)s * 16384, out, 1 << s);
    }
}
